// ChunkedAttention_87136296501295
// MI455X (gfx1250) — matrix operand order repaired and checked by reading, not compiled, not run on hardware
//
#include <hip/hip_runtime.h>
#include <hip/hip_bf16.h>
#include <stdint.h>

typedef _Float16 half_t;
typedef __attribute__((ext_vector_type(16))) _Float16 v16h;
typedef __attribute__((ext_vector_type(8)))  _Float16 v8h;
typedef __attribute__((ext_vector_type(8)))  float    v8f;
typedef __attribute__((ext_vector_type(4)))  unsigned int u32x4;
typedef __attribute__((ext_vector_type(8)))  int      i32x8;
typedef __attribute__((ext_vector_type(4)))  int      i32x4;

#define BM 128
#define BN 128
#define BK 32
#define LDA 40   // BK + 8 halves; row pitch 80B (16B-aligned)
#define LDB 40

#define NUM_HEADS 8
#define HEAD_DIM 128
#define CHUNK 64
#define SEQ_L 4096
#define CDIM 1024
#define BATCH 8
#define MROWS (BATCH * SEQ_L)

#if defined(__has_builtin)
#if __has_builtin(__builtin_amdgcn_tensor_load_to_lds)
#define USE_TDM 1
#endif
#endif
#ifndef USE_TDM
#define USE_TDM 0
#endif

__device__ __forceinline__ v8f zero8() {
  v8f z = {0.f, 0.f, 0.f, 0.f, 0.f, 0.f, 0.f, 0.f};
  return z;
}

__device__ __forceinline__ v8f wmma_f16(v16h a, v16h b, v8f c) {
  // D = A(16x32 f16) * B(32x16 f16) + C(16x16 f32)
  return __builtin_amdgcn_wmma_f32_16x16x32_f16(false, a, false, b, (short)0, c,
                                                false, false);
}

// Flat pointers to LDS carry the byte offset in addr[31:0] (LDS aperture).
__device__ __forceinline__ uint32_t lds_offset(const void* p) {
  return (uint32_t)(uintptr_t)p;
}

#if USE_TDM
// Issue a TDM 2D tile load (global -> LDS) per CDNA5 ISA.
// data_size = 2 bytes. pad_int: 0=2,1=4,2=8,3=16,4=32,5=64 DWORDs between pads.
// pad_amt: pad DWORDs - 1. Tracked on TENSORcnt.
__device__ __forceinline__ void tdm_load_tile(const void* gsrc, uint32_t lds_off,
                                              int tile_d0, int tile_d1,
                                              int tensor_d0, int tensor_d1,
                                              int stride0, int pad_int, int pad_amt) {
  const uint64_t ga = (uint64_t)(uintptr_t)gsrc;
  u32x4 g0;
  g0[0] = 1u;                                   // count=1, user descriptor
  g0[1] = lds_off;                              // lds_addr
  g0[2] = (uint32_t)(ga & 0xffffffffu);         // global_addr[31:0]
  g0[3] = (uint32_t)((ga >> 32) & 0x01ffffffu)  // global_addr[56:32]
          | (2u << 30);                         // type = 2 ("image")
  i32x8 g1;
  g1[0] = (int)((1u << 16)                      // data_size = 2 bytes
                | (1u << 20)                    // pad_enable
                | ((uint32_t)pad_int << 22)
                | ((uint32_t)pad_amt << 25));
  g1[1] = (int)(((uint32_t)tensor_d0 & 0xffffu) << 16);          // dim0[15:0]
  g1[2] = (int)((((uint32_t)tensor_d0 >> 16) & 0xffffu)
                | (((uint32_t)tensor_d1 & 0xffffu) << 16));      // dim0 hi | dim1 lo
  g1[3] = (int)((((uint32_t)tensor_d1 >> 16) & 0xffffu)
                | (((uint32_t)tile_d0 & 0xffffu) << 16));        // dim1 hi | tile0
  g1[4] = (int)((uint32_t)tile_d1 & 0xffffu);                    // tile1 | tile2=0
  g1[5] = stride0;                                               // dim0_stride[31:0]
  g1[6] = 0;                                                     // stride0 hi, stride1 lo
  g1[7] = 0;
  i32x4 z4 = {0, 0, 0, 0};                 // groups 2/3 unused for 2D tensors
  i32x8 z8 = {0, 0, 0, 0, 0, 0, 0, 0};
  __builtin_amdgcn_tensor_load_to_lds(g0, g1, z4, z4, z8, 0);
}
#endif

// A fragment (16x32, M x K) from LDS, f16 source.
__device__ __forceinline__ v16h load_frag_a(const half_t* s, int ld, int row_base,
                                            int k_base, int lane) {
  const int row = row_base + (lane & 15);
  const int kb  = k_base + ((lane >> 4) << 3);
  const half_t* p = s + row * ld + kb;
  v8h lo = *(const v8h*)p;          // K kb..kb+7   -> VGPRs 0..3
  v8h hi = *(const v8h*)(p + 16);   // K kb+16..+23 -> VGPRs 4..7
  v16h r;
#pragma unroll
  for (int i = 0; i < 8; ++i) { r[i] = lo[i]; r[i + 8] = hi[i]; }
  return r;
}

// Same A fragment but from an f32 LDS buffer (softmax probabilities).
__device__ __forceinline__ v16h load_frag_a_f32(const float* s, int ld, int row_base,
                                                int k_base, int lane) {
  const int row = row_base + (lane & 15);
  const int kb  = k_base + ((lane >> 4) << 3);
  const float* p = s + row * ld + kb;
  v16h r;
#pragma unroll
  for (int i = 0; i < 8; ++i) {
    r[i]     = (half_t)p[i];
    r[i + 8] = (half_t)p[i + 16];
  }
  return r;
}

// B fragment (32x16, K x N) from LDS holding B^T rows (K contiguous per row).
__device__ __forceinline__ v16h load_frag_b(const half_t* s, int ld, int col_base,
                                            int k_base, int lane) {
  const int col = col_base + (lane & 15);
  const int kb  = k_base + ((lane >> 4) << 3);
  const half_t* p = s + col * ld + kb;
  v8h lo = *(const v8h*)p;        // K kb..kb+7
  v8h hi = *(const v8h*)(p + 16); // K kb+16..kb+23
  v16h r;
#pragma unroll
  for (int i = 0; i < 8; ++i) { r[i] = lo[i]; r[i + 8] = hi[i]; }
  return r;
}

// ---------------------------------------------------------------------------
// Kernel 1: tiled transpose + f32->f16 convert.
// src viewed as [nb][R][Ccols] f32; dst[(b*Ccols + l)*R + r] = src[b][r][l].
// ---------------------------------------------------------------------------
__global__ __launch_bounds__(256) void transpose_convert_kernel(
    const float* __restrict__ src, half_t* __restrict__ dst, int R, int Ccols) {
  __shared__ float tile[32][33];
  const int b  = blockIdx.z;
  const int l0 = blockIdx.x * 32;
  const int r0 = blockIdx.y * 32;
  const int tx = threadIdx.x & 31;
  const int ty = threadIdx.x >> 5;  // 0..7
#pragma unroll
  for (int i = 0; i < 32; i += 8)
    tile[ty + i][tx] = src[((size_t)b * R + r0 + ty + i) * Ccols + l0 + tx];
  __syncthreads();
#pragma unroll
  for (int i = 0; i < 32; i += 8)
    dst[((size_t)b * Ccols + l0 + ty + i) * R + r0 + tx] = (half_t)tile[tx][ty + i];
}

// ---------------------------------------------------------------------------
// GEMM mainloop: acc += A[M,K] * Bt[N,K]^T for one 128x128 tile, 8 waves.
// TDM double-buffered tile feed: wave 0 issues tensor_load_to_lds for the
// next K-tile while all waves run WMMAs on the current one; wave 0 waits on
// TENSORcnt before the barrier so the barrier release implies DMA done.
// ---------------------------------------------------------------------------
__device__ __forceinline__ void gemm_nt_core(const half_t* __restrict__ A,
                                             const half_t* __restrict__ Bt,
                                             int M, int N, int K, int m0, int n0,
                                             int wm, int wn, int lane, int wave,
                                             int tid, v8f (&acc)[2][4]) {
  __shared__ half_t As[2][BM * LDA];
  __shared__ half_t Bs[2][BN * LDB];
#if USE_TDM
  if (wave == 0) {
    tdm_load_tile((const char*)A + (size_t)m0 * K * 2, lds_offset(As[0]),
                  BK, BM, K, M, K, 3, 3);
    tdm_load_tile((const char*)Bt + (size_t)n0 * K * 2, lds_offset(Bs[0]),
                  BK, BN, K, N, K, 3, 3);
    __builtin_amdgcn_s_wait_tensorcnt(0);
  }
  __syncthreads();
  for (int k0 = 0; k0 < K; k0 += BK) {
    const int buf = (k0 >> 5) & 1;
    const bool more = (k0 + BK) < K;
    if (wave == 0 && more) {
      tdm_load_tile((const char*)A + ((size_t)m0 * K + k0 + BK) * 2,
                    lds_offset(As[buf ^ 1]), BK, BM, K, M, K, 3, 3);
      tdm_load_tile((const char*)Bt + ((size_t)n0 * K + k0 + BK) * 2,
                    lds_offset(Bs[buf ^ 1]), BK, BN, K, N, K, 3, 3);
    }
    const half_t* Ac = As[buf];
    const half_t* Bc = Bs[buf];
    v16h af0 = load_frag_a(Ac, LDA, wm, 0, lane);
    v16h af1 = load_frag_a(Ac, LDA, wm + 16, 0, lane);
#pragma unroll
    for (int j = 0; j < 4; ++j) {
      v16h bf = load_frag_b(Bc, LDB, wn + j * 16, 0, lane);
      acc[0][j] = wmma_f16(af0, bf, acc[0][j]);
      acc[1][j] = wmma_f16(af1, bf, acc[1][j]);
    }
    if (wave == 0 && more) __builtin_amdgcn_s_wait_tensorcnt(0);
    __syncthreads();
  }
#else
  const int r0  = tid >> 2;        // 0..63
  const int c0v = (tid & 3) << 3;  // 0,8,16,24
  for (int k0 = 0; k0 < K; k0 += BK) {
    *(v8h*)&As[0][r0 * LDA + c0v] =
        *(const v8h*)&A[(size_t)(m0 + r0) * K + k0 + c0v];
    *(v8h*)&As[0][(r0 + 64) * LDA + c0v] =
        *(const v8h*)&A[(size_t)(m0 + r0 + 64) * K + k0 + c0v];
    *(v8h*)&Bs[0][r0 * LDB + c0v] =
        *(const v8h*)&Bt[(size_t)(n0 + r0) * K + k0 + c0v];
    *(v8h*)&Bs[0][(r0 + 64) * LDB + c0v] =
        *(const v8h*)&Bt[(size_t)(n0 + r0 + 64) * K + k0 + c0v];
    __syncthreads();
    v16h af0 = load_frag_a(As[0], LDA, wm, 0, lane);
    v16h af1 = load_frag_a(As[0], LDA, wm + 16, 0, lane);
#pragma unroll
    for (int j = 0; j < 4; ++j) {
      v16h bf = load_frag_b(Bs[0], LDB, wn + j * 16, 0, lane);
      acc[0][j] = wmma_f16(af0, bf, acc[0][j]);
      acc[1][j] = wmma_f16(af1, bf, acc[1][j]);
    }
    __syncthreads();
  }
#endif
}

// Y = A*Bt^T + bias, f16 output (Q/K/V projections).
__global__ __launch_bounds__(256, 2) void gemm_nt_f16_kernel(
    const half_t* __restrict__ A, const half_t* __restrict__ Bt,
    const float* __restrict__ bias, half_t* __restrict__ Y, int M, int N, int K) {
  const int tid  = threadIdx.x;
  const int lane = tid & 31;
  const int wave = tid >> 5;
  const int m0 = blockIdx.x * BM;
  const int n0 = blockIdx.y * BN;
  const int wm = (wave >> 1) * 32;
  const int wn = (wave & 1) * 64;
  v8f acc[2][4];
#pragma unroll
  for (int i = 0; i < 2; ++i)
#pragma unroll
    for (int j = 0; j < 4; ++j) acc[i][j] = zero8();

  gemm_nt_core(A, Bt, M, N, K, m0, n0, wm, wn, lane, wave, tid, acc);

  const int hi  = (lane >> 4) * 8;
  const int col = lane & 15;
#pragma unroll
  for (int i = 0; i < 2; ++i)
#pragma unroll
    for (int j = 0; j < 4; ++j) {
      const int n = n0 + wn + j * 16 + col;
      const float bv = bias[n];
      const size_t rowbase = (size_t)(m0 + wm + i * 16 + hi);
#pragma unroll
      for (int r = 0; r < 8; ++r)
        Y[(rowbase + r) * N + n] = (half_t)(acc[i][j][r] + bv);
    }
}

// Out[b, n, l] = (A*Bt^T + bias)[b*L + l, n], f32 (final projection).
// WMMA D layout gives each lane 8 consecutive M rows => two float4 stores.
__global__ __launch_bounds__(256, 2) void gemm_nt_f32t_kernel(
    const half_t* __restrict__ A, const half_t* __restrict__ Bt,
    const float* __restrict__ bias, float* __restrict__ Out, int M, int N, int K,
    int Lseq) {
  const int tid  = threadIdx.x;
  const int lane = tid & 31;
  const int wave = tid >> 5;
  const int m0 = blockIdx.x * BM;
  const int n0 = blockIdx.y * BN;
  const int wm = (wave >> 1) * 32;
  const int wn = (wave & 1) * 64;
  v8f acc[2][4];
#pragma unroll
  for (int i = 0; i < 2; ++i)
#pragma unroll
    for (int j = 0; j < 4; ++j) acc[i][j] = zero8();

  gemm_nt_core(A, Bt, M, N, K, m0, n0, wm, wn, lane, wave, tid, acc);

  const int hi  = (lane >> 4) * 8;
  const int col = lane & 15;
  const int bIdx = m0 / Lseq;  // BM divides Lseq
  const int lbase = m0 - bIdx * Lseq;
#pragma unroll
  for (int i = 0; i < 2; ++i)
#pragma unroll
    for (int j = 0; j < 4; ++j) {
      const int n = n0 + wn + j * 16 + col;
      const float bv = bias[n];
      const size_t base = (size_t)bIdx * N * Lseq + (size_t)n * Lseq +
                          (size_t)(lbase + wm + i * 16 + hi);
      float4 v0, v1;
      v0.x = acc[i][j][0] + bv; v0.y = acc[i][j][1] + bv;
      v0.z = acc[i][j][2] + bv; v0.w = acc[i][j][3] + bv;
      v1.x = acc[i][j][4] + bv; v1.y = acc[i][j][5] + bv;
      v1.z = acc[i][j][6] + bv; v1.w = acc[i][j][7] + bv;
      *(float4*)&Out[base]     = v0;
      *(float4*)&Out[base + 4] = v1;
    }
}

// ---------------------------------------------------------------------------
// Kernel 4: block-diagonal attention. One WG per (chunk, head).
// Q/K tiles via TDM; V fetched into registers up front (overlaps S-GEMM and
// softmax), scattered transposed into the dead Q/K LDS region afterwards.
// ---------------------------------------------------------------------------
__global__ __launch_bounds__(256, 2) void attn_chunk_kernel(
    const half_t* __restrict__ Qh, const half_t* __restrict__ Kh,
    const half_t* __restrict__ Vh, half_t* __restrict__ Oh) {
  __shared__ __align__(16) char smem[52224];
  half_t* Qs = (half_t*)smem;                 // [64][136] f16
  half_t* Ks = (half_t*)(smem + 17408);       // [64][136] f16
  half_t* Vt = (half_t*)smem;                 // [128][72] f16, aliases Qs/Ks
  float*  Ss = (float*)(smem + 34816);        // [64][68]  f32

  const int tid  = threadIdx.x;
  const int lane = tid & 31;
  const int wave = tid >> 5;
  const int h  = blockIdx.y;
  const size_t q0 = (size_t)blockIdx.x * CHUNK;  // token base in [B*L]
  const int hoff = h * HEAD_DIM;

  // V chunk -> registers now; consumed after softmax.
  v8h vreg[4];
#pragma unroll
  for (int u = 0; u < 4; ++u) {
    const int i = tid + u * 256;
    const int r = i >> 4;         // token
    const int c = (i & 15) << 3;  // dim base
    vreg[u] = *(const v8h*)&Vh[(q0 + r) * CDIM + hoff + c];
  }

#if USE_TDM
  if (wave == 0) {
    // 64x128-half tiles, row = 64 DWORDs (pad_int=5), pad 4 DWORDs (pad_amt=3)
    tdm_load_tile(&Qh[q0 * CDIM + hoff], lds_offset(Qs), HEAD_DIM, CHUNK,
                  CDIM, MROWS, CDIM, 5, 3);
    tdm_load_tile(&Kh[q0 * CDIM + hoff], lds_offset(Ks), HEAD_DIM, CHUNK,
                  CDIM, MROWS, CDIM, 5, 3);
    __builtin_amdgcn_s_wait_tensorcnt(0);
  }
#else
#pragma unroll
  for (int u = 0; u < 4; ++u) {
    const int i = tid + u * 256;
    const int r = i >> 4;
    const int c = (i & 15) << 3;
    *(v8h*)&Qs[r * 136 + c] = *(const v8h*)&Qh[(q0 + r) * CDIM + hoff + c];
    *(v8h*)&Ks[r * 136 + c] = *(const v8h*)&Kh[(q0 + r) * CDIM + hoff + c];
  }
#endif
  __syncthreads();

  // S = Q K^T : 4x4 grid of 16x16 tiles, 2 per wave.
  const int mt = wave >> 1;
  const int nh = wave & 1;
  v8f sacc[2] = {zero8(), zero8()};
#pragma unroll
  for (int k = 0; k < HEAD_DIM; k += 32) {
    v16h a = load_frag_a(Qs, 136, mt * 16, k, lane);
#pragma unroll
    for (int j = 0; j < 2; ++j) {
      v16h b = load_frag_b(Ks, 136, (nh * 2 + j) * 16, k, lane);
      sacc[j] = wmma_f16(a, b, sacc[j]);
    }
  }
  const float scale = 0.08838834764831845f;  // 1/sqrt(128)
  const int hi  = (lane >> 4) * 8;
  const int col = lane & 15;
#pragma unroll
  for (int j = 0; j < 2; ++j)
#pragma unroll
    for (int r = 0; r < 8; ++r)
      Ss[(mt * 16 + hi + r) * 68 + (nh * 2 + j) * 16 + col] = sacc[j][r] * scale;
  __syncthreads();  // Q/K reads complete; Qs/Ks region now dead

  // Scatter V (transposed) into Vt while doing softmax on Ss.
#pragma unroll
  for (int u = 0; u < 4; ++u) {
    const int i = tid + u * 256;
    const int r = i >> 4;
    const int c = (i & 15) << 3;
#pragma unroll
    for (int e = 0; e < 8; ++e) Vt[(c + e) * 72 + r] = vreg[u][e];
  }
  // Softmax: wave w owns rows 8w..8w+7; 2 elems/lane, shfl_xor (wave32).
#pragma unroll
  for (int rr = 0; rr < 8; ++rr) {
    float* row = &Ss[(wave * 8 + rr) * 68];
    float a  = row[lane];
    float b2 = row[lane + 32];
    float mx = fmaxf(a, b2);
#pragma unroll
    for (int off = 16; off > 0; off >>= 1) mx = fmaxf(mx, __shfl_xor(mx, off));
    const float e0 = __expf(a - mx);
    const float e1 = __expf(b2 - mx);
    float s = e0 + e1;
#pragma unroll
    for (int off = 16; off > 0; off >>= 1) s += __shfl_xor(s, off);
    const float inv = 1.0f / s;
    row[lane]      = e0 * inv;
    row[lane + 32] = e1 * inv;
  }
  __syncthreads();

  // O = P V : 4x8 grid of 16x16 tiles, 4 per wave; P read from f32 LDS.
  v8f oacc[4] = {zero8(), zero8(), zero8(), zero8()};
  const int ntb = nh * 4;
#pragma unroll
  for (int k = 0; k < CHUNK; k += 32) {
    v16h a = load_frag_a_f32(Ss, 68, mt * 16, k, lane);
#pragma unroll
    for (int j = 0; j < 4; ++j) {
      v16h b = load_frag_b(Vt, 72, (ntb + j) * 16, k, lane);
      oacc[j] = wmma_f16(a, b, oacc[j]);
    }
  }
#pragma unroll
  for (int j = 0; j < 4; ++j)
#pragma unroll
    for (int r = 0; r < 8; ++r) {
      const int m = mt * 16 + hi + r;
      const int n = (ntb + j) * 16 + col;
      Oh[(q0 + m) * CDIM + hoff + n] = (half_t)oacc[j][r];
    }
}

// ---------------------------------------------------------------------------
extern "C" void kernel_launch(void* const* d_in, const int* in_sizes, int n_in,
                              void* d_out, int out_size, void* d_ws, size_t ws_size,
                              hipStream_t stream) {
  const float* x  = (const float*)d_in[0];
  const float* Wq = (const float*)d_in[1];
  const float* bq = (const float*)d_in[2];
  const float* Wk = (const float*)d_in[3];
  const float* bk = (const float*)d_in[4];
  const float* Wv = (const float*)d_in[5];
  const float* bv = (const float*)d_in[6];
  const float* Wo = (const float*)d_in[7];
  const float* bo = (const float*)d_in[8];
  float* out = (float*)d_out;

  char* ws = (char*)d_ws;
  const size_t MB = 1024ull * 1024ull;
  half_t* xh  = (half_t*)ws;               // 64 MB, later reused as attention out
  half_t* Wtq = (half_t*)(ws + 64 * MB);   // 2 MB each
  half_t* Wtk = (half_t*)(ws + 66 * MB);
  half_t* Wtv = (half_t*)(ws + 68 * MB);
  half_t* Wto = (half_t*)(ws + 70 * MB);
  half_t* Qh  = (half_t*)(ws + 72 * MB);   // 64 MB
  half_t* Kh  = (half_t*)(ws + 136 * MB);  // 64 MB
  half_t* Vh  = (half_t*)(ws + 200 * MB);  // 64 MB

  // 1) x[B,C,L] f32 -> xh[B*L, C] f16 (transpose+convert)
  transpose_convert_kernel<<<dim3(SEQ_L / 32, CDIM / 32, BATCH), 256, 0, stream>>>(
      x, xh, CDIM, SEQ_L);
  // 2) W[k,n] f32 -> Wt[n,k] f16 for NT GEMM
  dim3 wg(CDIM / 32, CDIM / 32, 1);
  transpose_convert_kernel<<<wg, 256, 0, stream>>>(Wq, Wtq, CDIM, CDIM);
  transpose_convert_kernel<<<wg, 256, 0, stream>>>(Wk, Wtk, CDIM, CDIM);
  transpose_convert_kernel<<<wg, 256, 0, stream>>>(Wv, Wtv, CDIM, CDIM);
  transpose_convert_kernel<<<wg, 256, 0, stream>>>(Wo, Wto, CDIM, CDIM);

  // 3) Q/K/V projections (WMMA GEMMs, TDM tile feed)
  dim3 gg(MROWS / BM, CDIM / BN);
  gemm_nt_f16_kernel<<<gg, 256, 0, stream>>>(xh, Wtq, bq, Qh, MROWS, CDIM, CDIM);
  gemm_nt_f16_kernel<<<gg, 256, 0, stream>>>(xh, Wtk, bk, Kh, MROWS, CDIM, CDIM);
  gemm_nt_f16_kernel<<<gg, 256, 0, stream>>>(xh, Wtv, bv, Vh, MROWS, CDIM, CDIM);

  // 4) Block-diagonal attention; output reuses xh buffer
  attn_chunk_kernel<<<dim3(MROWS / CHUNK, NUM_HEADS), 256, 0, stream>>>(Qh, Kh, Vh, xh);

  // 5) Output projection with fused transpose to out[B, C, L] f32
  gemm_nt_f32t_kernel<<<gg, 256, 0, stream>>>(xh, Wto, bo, out, MROWS, CDIM, CDIM,
                                              SEQ_L);
}
